// Block_29927332118937
// MI455X (gfx1250) — compile-verified, matrix operand order checked by reading, not run on hardware
//
#include <hip/hip_runtime.h>
#include <hip/hip_bf16.h>
#include <math.h>
#include <stdint.h>

// ---------------------------------------------------------------------------
// Types for CDNA5 WMMA (wave32): 16x16x32 f16 -> f32
// ---------------------------------------------------------------------------
typedef _Float16 v16h __attribute__((ext_vector_type(16)));
typedef _Float16 v8h  __attribute__((ext_vector_type(8)));
typedef float    v8f  __attribute__((ext_vector_type(8)));

static __device__ __forceinline__ v8f wmma_f16(v16h a, v16h b, v8f c) {
  // (neg_a, A, neg_b, B, c_mod, C, reuse_a, reuse_b)
  return __builtin_amdgcn_wmma_f32_16x16x32_f16(false, a, false, b, (short)0, c,
                                                false, false);
}

static __device__ __forceinline__ v8f zero8() {
  v8f z;
#pragma unroll
  for (int i = 0; i < 8; ++i) z[i] = 0.0f;
  return z;
}

// Build a 16x32 f16 fragment (documented CDNA5 A/B layout) from a per-lane
// pointer to 8+8 contiguous halves:
//   lane L(0..15):  row rc0+L,    elems 0..7 -> k0+0..7,  8..15 -> k0+16..23
//   lane L(16..31): row rc0+L-16, elems 0..7 -> k0+8..15, 8..15 -> k0+24..31
static __device__ __forceinline__ v16h frag_from(const _Float16* p) {
  v8h lo = *(const v8h*)p;
  v8h hi = *(const v8h*)(p + 16);
  return __builtin_shufflevector(lo, hi, 0, 1, 2, 3, 4, 5, 6, 7, 8, 9, 10, 11,
                                 12, 13, 14, 15);
}

// Global-memory fragment load (row-major, leading dim ld).
static __device__ __forceinline__ v16h load_frag(const _Float16* __restrict__ base,
                                                 int ld, int rc0, int k0) {
  const int lane = threadIdx.x & 31;
  return frag_from(base + (size_t)(rc0 + (lane & 15)) * ld +
                   (k0 + ((lane >> 4) << 3)));
}

// Async global->LDS 128-bit copy (CDNA5, ASYNCcnt-tracked).
static __device__ __forceinline__ void async_copy_b128(const _Float16* gsrc,
                                                       _Float16* ldst) {
  unsigned loff = (unsigned)(uintptr_t)ldst;  // low 32 bits = LDS offset
  asm volatile("global_load_async_to_lds_b128 %0, %1, off" ::"v"(loff),
               "v"((uint64_t)(uintptr_t)gsrc)
               : "memory");
}

static __device__ __forceinline__ void wait_async_all() {
  asm volatile("s_wait_asynccnt 0" ::: "memory");
}

// ---------------------------------------------------------------------------
// LayerNorm (population std, eps on std) fp32 -> f16.  One 256-thr block/row.
// ---------------------------------------------------------------------------
__global__ __launch_bounds__(256) void ln_kernel(const float* __restrict__ x,
                                                 const float* __restrict__ w,
                                                 const float* __restrict__ bias,
                                                 _Float16* __restrict__ out,
                                                 int C) {
  const int row = blockIdx.x;
  const float* xr = x + (size_t)row * C;
  __shared__ float red[256];
  float local[4];
  float s = 0.0f;
#pragma unroll
  for (int i = 0; i < 4; ++i) {
    local[i] = xr[threadIdx.x + 256 * i];
    s += local[i];
  }
  red[threadIdx.x] = s;
  __syncthreads();
  for (int off = 128; off > 0; off >>= 1) {
    if (threadIdx.x < off) red[threadIdx.x] += red[threadIdx.x + off];
    __syncthreads();
  }
  const float mean = red[0] / (float)C;
  __syncthreads();
  s = 0.0f;
#pragma unroll
  for (int i = 0; i < 4; ++i) {
    float d = local[i] - mean;
    s += d * d;
  }
  red[threadIdx.x] = s;
  __syncthreads();
  for (int off = 128; off > 0; off >>= 1) {
    if (threadIdx.x < off) red[threadIdx.x] += red[threadIdx.x + off];
    __syncthreads();
  }
  const float inv = 1.0f / (sqrtf(red[0] / (float)C) + 1e-5f);
#pragma unroll
  for (int i = 0; i < 4; ++i) {
    int c = threadIdx.x + 256 * i;
    out[(size_t)row * C + c] =
        (_Float16)((local[i] - mean) * inv * w[c] + bias[c]);
  }
}

// ---------------------------------------------------------------------------
// fp32 W[K][N]  ->  f16 Wᵀ[N][K]  (so GEMM B-fragments are contiguous loads)
// ---------------------------------------------------------------------------
__global__ __launch_bounds__(256) void convertT_kernel(
    const float* __restrict__ W, _Float16* __restrict__ WT, int K, int N) {
  size_t idx = (size_t)blockIdx.x * blockDim.x + threadIdx.x;
  if (idx >= (size_t)K * N) return;
  int k = (int)(idx % K);
  int n = (int)(idx / K);
  WT[idx] = (_Float16)W[(size_t)k * N + n];
}

// ---------------------------------------------------------------------------
// WMMA GEMM: C[M][N] = A[M][K] (f16, row-major) x Bᵀ[N][K] (f16) + bias
//   8 waves / block, each wave a 64x32 tile, block tile 128x128.
//   K-step 32, double-buffered LDS staged via async global->LDS copies.
// Epilogues: 0 = f16 out, 1 = f32 out + residual add, 2 = f16 out + GELU.
// ---------------------------------------------------------------------------
enum { EPI_H16 = 0, EPI_F32_RES = 1, EPI_H16_GELU = 2 };

#define LDSPITCH 40  // 128 rows x (32 + 8 pad) halves; 20-bank row stride

template <int EPI>
__global__ __launch_bounds__(256) void gemm_kernel(
    const _Float16* __restrict__ A, const _Float16* __restrict__ BT,
    const float* __restrict__ bias, const float* __restrict__ resid,
    float* __restrict__ outF, _Float16* __restrict__ outH, int M, int N,
    int K) {
  __shared__ _Float16 sA[2][128 * LDSPITCH];
  __shared__ _Float16 sB[2][128 * LDSPITCH];
  const int tid = threadIdx.x;
  const int wave = tid >> 5;
  const int lane = tid & 31;
  const int rowblk = blockIdx.y * 128;
  const int colblk = blockIdx.x * 128;
  const int wrow = (wave & 1) * 64;
  const int wcol = (wave >> 1) * 32;

  // Stage a 128x32 A tile + 128x32 B tile into buffer `buf` for K offset k0.
  // 256 threads x 2 chunks x 16B each per matrix.
  auto stage = [&](int buf, int k0) {
#pragma unroll
    for (int i = 0; i < 2; ++i) {
      const int c = tid + 256 * i;     // chunk id 0..511
      const int row = c >> 2;          // 0..127
      const int ko = (c & 3) * 8;      // 0,8,16,24 halves
      async_copy_b128(A + (size_t)(rowblk + row) * K + k0 + ko,
                      &sA[buf][row * LDSPITCH + ko]);
      async_copy_b128(BT + (size_t)(colblk + row) * K + k0 + ko,
                      &sB[buf][row * LDSPITCH + ko]);
    }
  };

  v8f acc[4][2];
#pragma unroll
  for (int i = 0; i < 4; ++i)
#pragma unroll
    for (int j = 0; j < 2; ++j) acc[i][j] = zero8();

  stage(0, 0);
  wait_async_all();
  __syncthreads();

  int buf = 0;
  const int kh = (lane >> 4) << 3;
  const int lr = lane & 15;
  for (int k0 = 0; k0 < K; k0 += 32) {
    if (k0 + 32 < K) stage(buf ^ 1, k0 + 32);  // copy engine runs ahead

    v16h a[4], b[2];
#pragma unroll
    for (int i = 0; i < 4; ++i)
      a[i] = frag_from(&sA[buf][(wrow + 16 * i + lr) * LDSPITCH + kh]);
#pragma unroll
    for (int j = 0; j < 2; ++j)
      b[j] = frag_from(&sB[buf][(wcol + 16 * j + lr) * LDSPITCH + kh]);
#pragma unroll
    for (int i = 0; i < 4; ++i)
#pragma unroll
      for (int j = 0; j < 2; ++j) acc[i][j] = wmma_f16(a[i], b[j], acc[i][j]);

    wait_async_all();   // own async copies for next buffer done
    __syncthreads();    // ... and everyone else's
    buf ^= 1;
  }

  // C/D layout: elem r of lane L -> row = r + (L>=16)*8, col = L%16
  const int rofs = (lane >> 4) << 3;
#pragma unroll
  for (int i = 0; i < 4; ++i)
#pragma unroll
    for (int j = 0; j < 2; ++j)
#pragma unroll
      for (int r = 0; r < 8; ++r) {
        const int row = rowblk + wrow + 16 * i + r + rofs;
        const int col = colblk + wcol + 16 * j + lr;
        float v = acc[i][j][r] + bias[col];
        const size_t o = (size_t)row * N + col;
        if (EPI == EPI_H16) {
          outH[o] = (_Float16)v;
        } else if (EPI == EPI_F32_RES) {
          outF[o] = resid[o] + v;
        } else {  // exact GELU: 0.5 x (1 + erf(x/sqrt(2)))
          outH[o] = (_Float16)(0.5f * v * (1.0f + erff(v * 0.70710678f)));
        }
      }
}

// ---------------------------------------------------------------------------
// qkv f16 [B*T][3C] -> Q[B,H,T,D], K[B,H,T,D], Vᵀ[B,H,D,T]  (all f16)
// ---------------------------------------------------------------------------
__global__ __launch_bounds__(256) void pack_qkv_kernel(
    const _Float16* __restrict__ qkv, _Float16* __restrict__ Qh,
    _Float16* __restrict__ Kh, _Float16* __restrict__ VTh, int Bn, int T,
    int C, int H, int D) {
  size_t idx = (size_t)blockIdx.x * blockDim.x + threadIdx.x;
  if (idx >= (size_t)Bn * T * C) return;
  int c = (int)(idx % C);
  size_t bt = idx / C;
  int t = (int)(bt % T);
  int b = (int)(bt / T);
  int h = c / D, d = c % D;
  const _Float16* src = qkv + bt * (size_t)(3 * C);
  size_t o = ((size_t)(b * H + h) * T + t) * D + d;
  Qh[o] = src[c];
  Kh[o] = src[C + c];
  VTh[((size_t)(b * H + h) * D + d) * T + t] = src[2 * C + c];
}

// ---------------------------------------------------------------------------
// Causal flash attention. One wave per 16-query tile per (b,h).
// S = QKᵀ and O += P·V both on WMMA; online softmax with 16-lane xor
// reductions; C-layout -> A-layout hop for P through per-wave LDS.
// ---------------------------------------------------------------------------
__global__ __launch_bounds__(256) void attn_kernel(
    const _Float16* __restrict__ Q, const _Float16* __restrict__ Kt,
    const _Float16* __restrict__ VT, _Float16* __restrict__ Y, int H, int T,
    int D, int C) {
  __shared__ _Float16 sP[8][16 * 34];  // stride 34 halfs: bank-conflict-free-ish
  const int wave = threadIdx.x >> 5;
  const int lane = threadIdx.x & 31;
  const int gid = blockIdx.x * 8 + wave;
  const int qtiles = T >> 4;
  const int bh = gid / qtiles;
  const int qt = (gid % qtiles) << 4;
  const int b = bh / H, h = bh % H;
  const _Float16* Qb = Q + (size_t)bh * T * D;
  const _Float16* Kb = Kt + (size_t)bh * T * D;
  const _Float16* Vb = VT + (size_t)bh * D * T;
  _Float16* sp = &sP[wave][0];

  const int hf = lane >> 4;  // lane half (row-group select)
  const int lc = lane & 15;  // column lane
  const float sc = 0.125f;   // 1/sqrt(D), D = 64

  const v16h qa0 = load_frag(Qb, D, qt, 0);
  const v16h qa1 = load_frag(Qb, D, qt, 32);

  v8f acc[4];
#pragma unroll
  for (int j = 0; j < 4; ++j) acc[j] = zero8();
  float m[8], l[8];
#pragma unroll
  for (int r = 0; r < 8; ++r) {
    m[r] = -1e30f;
    l[r] = 0.0f;
  }

  for (int kt = 0; kt <= qt; kt += 32) {
    asm volatile("" ::: "memory");  // keep this iter's LDS stores after prior loads
    // --- S = Q·Kᵀ for 32 keys (two 16-key column tiles, K-dim = D = 64) ---
    v8f s0 = zero8(), s1 = zero8();
    s0 = wmma_f16(qa0, load_frag(Kb, D, kt, 0), s0);
    s0 = wmma_f16(qa1, load_frag(Kb, D, kt, 32), s0);
    s1 = wmma_f16(qa0, load_frag(Kb, D, kt + 16, 0), s1);
    s1 = wmma_f16(qa1, load_frag(Kb, D, kt + 16, 32), s1);

#pragma unroll
    for (int r = 0; r < 8; ++r) {
      const int q = qt + r + (hf << 3);
      float v0 = (kt + lc <= q) ? s0[r] * sc : -1e30f;
      float v1 = (kt + 16 + lc <= q) ? s1[r] * sc : -1e30f;
      // row max across the 16 lanes of this half-wave
      float t = fmaxf(v0, v1);
      t = fmaxf(t, __shfl_xor(t, 1, 32));
      t = fmaxf(t, __shfl_xor(t, 2, 32));
      t = fmaxf(t, __shfl_xor(t, 4, 32));
      t = fmaxf(t, __shfl_xor(t, 8, 32));
      const float mnew = fmaxf(m[r], t);
      const float corr = __expf(m[r] - mnew);
      const float p0 = __expf(v0 - mnew);
      const float p1 = __expf(v1 - mnew);
      float rs = p0 + p1;
      rs += __shfl_xor(rs, 1, 32);
      rs += __shfl_xor(rs, 2, 32);
      rs += __shfl_xor(rs, 4, 32);
      rs += __shfl_xor(rs, 8, 32);
      l[r] = l[r] * corr + rs;
      m[r] = mnew;
      acc[0][r] *= corr;
      acc[1][r] *= corr;
      acc[2][r] *= corr;
      acc[3][r] *= corr;
      const int prow = r + (hf << 3);
      sp[prow * 34 + lc] = (_Float16)p0;
      sp[prow * 34 + 16 + lc] = (_Float16)p1;
    }
    // LDS RAW across lanes within one wave: DS ops are in-order per wave,
    // enforce completion + forbid compiler reordering.
    asm volatile("s_wait_dscnt 0" ::: "memory");
    // Re-read P in A-fragment layout.
    v16h pa;
#pragma unroll
    for (int i = 0; i < 8; ++i) {
      pa[i] = sp[lc * 34 + (hf << 3) + i];
      pa[8 + i] = sp[lc * 34 + 16 + (hf << 3) + i];
    }
    // --- O += P·V ; Vᵀ[D][T] makes B-fragments contiguous in the key dim ---
#pragma unroll
    for (int j = 0; j < 4; ++j)
      acc[j] = wmma_f16(pa, load_frag(Vb, T, j * 16, kt), acc[j]);
  }

  // Normalize and scatter to y (f16 [B*T][C]) for the projection GEMM.
#pragma unroll
  for (int j = 0; j < 4; ++j)
#pragma unroll
    for (int r = 0; r < 8; ++r) {
      const int q = qt + r + (hf << 3);
      Y[(size_t)(b * T + q) * C + h * D + j * 16 + lc] =
          (_Float16)(acc[j][r] / l[r]);
    }
}

// ---------------------------------------------------------------------------
// Host-side orchestration
// ---------------------------------------------------------------------------
extern "C" void kernel_launch(void* const* d_in, const int* in_sizes, int n_in,
                              void* d_out, int out_size, void* d_ws,
                              size_t ws_size, hipStream_t stream) {
  (void)in_sizes; (void)n_in; (void)out_size; (void)ws_size;
  const int B = 4, T = 1024, C = 1024, H = 16, D = 64;
  const int M = B * T;  // 4096

  const float* x        = (const float*)d_in[0];
  const float* c_attn_w = (const float*)d_in[1];
  const float* c_attn_b = (const float*)d_in[2];
  const float* c_proj_w = (const float*)d_in[3];
  const float* c_proj_b = (const float*)d_in[4];
  const float* ln1_w    = (const float*)d_in[5];
  const float* ln1_b    = (const float*)d_in[6];
  const float* ln2_w    = (const float*)d_in[7];
  const float* ln2_b    = (const float*)d_in[8];
  const float* fc_w     = (const float*)d_in[9];
  const float* fc_b     = (const float*)d_in[10];
  const float* proj_w   = (const float*)d_in[11];
  const float* proj_b   = (const float*)d_in[12];
  float* out = (float*)d_out;

  // Workspace carve-up (~143 MB total), 256B aligned.
  char* ws = (char*)d_ws;
  size_t off = 0;
  auto alloc = [&](size_t bytes) -> char* {
    char* p = ws + off;
    off += (bytes + 255) & ~(size_t)255;
    return p;
  };
  _Float16* h1H     = (_Float16*)alloc((size_t)M * C * 2);          // ln1(x)
  _Float16* wT_attn = (_Float16*)alloc((size_t)3 * C * C * 2);      // [3C][C]
  _Float16* qkvH    = (_Float16*)alloc((size_t)M * 3 * C * 2);
  _Float16* Qh      = (_Float16*)alloc((size_t)M * C * 2);
  _Float16* Kh      = (_Float16*)alloc((size_t)M * C * 2);
  _Float16* VTh     = (_Float16*)alloc((size_t)M * C * 2);
  _Float16* yh      = (_Float16*)alloc((size_t)M * C * 2);
  _Float16* wT_proj = (_Float16*)alloc((size_t)C * C * 2);
  float*    x1      = (float*)   alloc((size_t)M * C * 4);          // x + attn
  _Float16* h2H     = (_Float16*)alloc((size_t)M * C * 2);          // ln2(x1)
  _Float16* wT_fc   = (_Float16*)alloc((size_t)4 * C * C * 2);      // [4C][C]
  _Float16* hgH     = (_Float16*)alloc((size_t)M * 4 * C * 2);
  _Float16* wT_pr2  = (_Float16*)alloc((size_t)C * 4 * C * 2);      // [C][4C]

  auto cdiv = [](size_t a, size_t b) { return (unsigned)((a + b - 1) / b); };

  // 1) Weight transpose+convert to f16 (N-major for WMMA B fragments)
  convertT_kernel<<<cdiv((size_t)C * 3 * C, 256), 256, 0, stream>>>(
      c_attn_w, wT_attn, C, 3 * C);
  convertT_kernel<<<cdiv((size_t)C * C, 256), 256, 0, stream>>>(
      c_proj_w, wT_proj, C, C);
  convertT_kernel<<<cdiv((size_t)C * 4 * C, 256), 256, 0, stream>>>(
      fc_w, wT_fc, C, 4 * C);
  convertT_kernel<<<cdiv((size_t)4 * C * C, 256), 256, 0, stream>>>(
      proj_w, wT_pr2, 4 * C, C);

  // 2) ln1 -> h1 (f16)
  ln_kernel<<<M, 256, 0, stream>>>(x, ln1_w, ln1_b, h1H, C);

  // 3) qkv = h1 @ c_attn_w + b  (f16 out)
  gemm_kernel<EPI_H16><<<dim3(3 * C / 128, M / 128), 256, 0, stream>>>(
      h1H, wT_attn, c_attn_b, nullptr, nullptr, qkvH, M, 3 * C, C);

  // 4) split heads; V transposed to [B,H,D,T]
  pack_qkv_kernel<<<cdiv((size_t)M * C, 256), 256, 0, stream>>>(
      qkvH, Qh, Kh, VTh, B, T, C, H, D);

  // 5) causal attention -> y (f16 [B*T][C])
  attn_kernel<<<(B * H * (T / 16)) / 8, 256, 0, stream>>>(Qh, Kh, VTh, yh, H,
                                                          T, D, C);

  // 6) x1 = x + y @ c_proj_w + b  (f32)
  gemm_kernel<EPI_F32_RES><<<dim3(C / 128, M / 128), 256, 0, stream>>>(
      yh, wT_proj, c_proj_b, x, x1, nullptr, M, C, C);

  // 7) ln2 -> h2 (f16)
  ln_kernel<<<M, 256, 0, stream>>>(x1, ln2_w, ln2_b, h2H, C);

  // 8) hg = gelu(h2 @ fc_w + b)  (f16)
  gemm_kernel<EPI_H16_GELU><<<dim3(4 * C / 128, M / 128), 256, 0, stream>>>(
      h2H, wT_fc, fc_b, nullptr, nullptr, hgH, M, 4 * C, C);

  // 9) out = x1 + hg @ proj_w + b  (f32)
  gemm_kernel<EPI_F32_RES><<<dim3(C / 128, M / 128), 256, 0, stream>>>(
      hgH, wT_pr2, proj_b, x1, out, nullptr, M, C, 4 * C);
}
